// RBLNGptOssMLP_77111842832398
// MI455X (gfx1250) — compile-verified, matrix operand order checked by reading, not run on hardware
//
#include <hip/hip_runtime.h>
#include <hip/hip_bf16.h>
#include <stdint.h>

#define E_ 16
#define H_ 2048
#define I_ 2048
#define K_ 4
#define T_ 1024
#define NBLK 64               // H_/32 (= I_/32)
#define ALPHA_ 1.702f
#define LIMIT_ 7.0f

typedef __attribute__((ext_vector_type(16))) __bf16   v16bf;
typedef __attribute__((ext_vector_type(8)))  float    v8f;
typedef __attribute__((ext_vector_type(4)))  unsigned u32x4;
typedef __attribute__((ext_vector_type(8)))  int      i32x8;
typedef __attribute__((ext_vector_type(4)))  int      i32x4;

union ABfrag { v16bf bf; unsigned u[8]; uint4 q[2]; };

// Single-instruction bf16 pair pack: D[15:0]=bf16(lo), D[31:16]=bf16(hi).
__device__ __forceinline__ unsigned cvt_pk_bf16(float lo, float hi) {
  unsigned d;
  asm("v_cvt_pk_bf16_f32 %0, %1, %2" : "=v"(d) : "v"(lo), "v"(hi));
  return d;
}

// B fragment (32x16 bf16) for one 16-wide N tile: this lane's weight row,
// dequantized by per-block scale. K pairs for lane-half kh are two runs of 8
// contiguous elements -> four b128 loads.
__device__ __forceinline__ void load_b_frag(const float* rowk, int kh, float s,
                                            ABfrag& b) {
  float4 w0 = *(const float4*)(rowk + 8 * kh);
  float4 w1 = *(const float4*)(rowk + 8 * kh + 4);
  float4 w2 = *(const float4*)(rowk + 16 + 8 * kh);
  float4 w3 = *(const float4*)(rowk + 16 + 8 * kh + 4);
  b.u[0] = cvt_pk_bf16(w0.x * s, w0.y * s);
  b.u[1] = cvt_pk_bf16(w0.z * s, w0.w * s);
  b.u[2] = cvt_pk_bf16(w1.x * s, w1.y * s);
  b.u[3] = cvt_pk_bf16(w1.z * s, w1.w * s);
  b.u[4] = cvt_pk_bf16(w2.x * s, w2.y * s);
  b.u[5] = cvt_pk_bf16(w2.z * s, w2.w * s);
  b.u[6] = cvt_pk_bf16(w3.x * s, w3.y * s);
  b.u[7] = cvt_pk_bf16(w3.z * s, w3.w * s);
}

// Swizzled A tile layout (16 rows x 2048 bf16 = 64 KB):
//   16-byte group g of row m lives at group index (g ^ m) within the row.
// Row stride is 4096 B (bank-aligned), but the XOR spreads the 16 lanes of a
// half-wave across 16 distinct 4-bank spans -> conflict-free ds_load_b128.
__device__ __forceinline__ void load_a_frag_sw(const uint4* abase, int m,
                                               int k0, int kh, ABfrag& a) {
  int g0 = (k0 >> 3) + kh;                 // 16B group of first 8-elem run
  a.q[0] = abase[(m << 8) + ((g0)     ^ m)];
  a.q[1] = abase[(m << 8) + ((g0 + 2) ^ m)];   // second run: +16 elems
}

// Swizzled element offset of (row m, element k) inside a 16x2048 tile.
__device__ __forceinline__ int swz_elem(int m, int k) {
  int D = k >> 1, g = D >> 2, w = D & 3;
  return (m << 11) + (((((g ^ m) << 2) | w) << 1) | (k & 1));
}

// ---------------------------------------------------------------- router ----
__global__ __launch_bounds__(256) void router_kernel(
    const float* __restrict__ x, const float* __restrict__ rw,
    const float* __restrict__ rb, float* __restrict__ scores) {
  int wave = threadIdx.x >> 5, lane = threadIdx.x & 31;
  int t = blockIdx.x * 8 + wave;
  float acc[E_];
#pragma unroll
  for (int e = 0; e < E_; ++e) acc[e] = 0.f;
  for (int h = lane; h < H_; h += 32) {
    float xv = x[(size_t)t * H_ + h];
#pragma unroll
    for (int e = 0; e < E_; ++e) acc[e] += xv * rw[(size_t)e * H_ + h];
  }
#pragma unroll
  for (int e = 0; e < E_; ++e)
    for (int off = 16; off; off >>= 1) acc[e] += __shfl_xor(acc[e], off, 32);
  if (lane == 0) {
    float v[E_];
#pragma unroll
    for (int e = 0; e < E_; ++e) v[e] = acc[e] + rb[e];
    int bi[K_]; float bv[K_]; bool used[E_];
#pragma unroll
    for (int e = 0; e < E_; ++e) used[e] = false;
    for (int k = 0; k < K_; ++k) {
      float best = -3.4e38f; int bidx = 0;
      for (int e = 0; e < E_; ++e)
        if (!used[e] && v[e] > best) { best = v[e]; bidx = e; }
      used[bidx] = true; bv[k] = best; bi[k] = bidx;
    }
    float mx = bv[0], sum = 0.f, ex[K_];
    for (int k = 0; k < K_; ++k) { ex[k] = __expf(bv[k] - mx); sum += ex[k]; }
    float o[E_];
#pragma unroll
    for (int e = 0; e < E_; ++e) o[e] = 0.f;
    for (int k = 0; k < K_; ++k) o[bi[k]] = ex[k] / sum;
    for (int e = 0; e < E_; ++e) scores[(size_t)t * E_ + e] = o[e];
  }
}

// --------------------------------------------------------------- compact ----
__global__ void compact_kernel(const float* __restrict__ scores,
                               int* __restrict__ cnt, int* __restrict__ tokidx,
                               int* __restrict__ slotof) {
  int e = threadIdx.x;
  if (e >= E_) return;
  int c = 0;
  for (int t = 0; t < T_; ++t) {
    if (scores[(size_t)t * E_ + e] > 0.f) {
      tokidx[e * T_ + c] = t;
      slotof[t * E_ + e] = c;
      ++c;
    } else {
      slotof[t * E_ + e] = 0;
    }
  }
  cnt[e] = c;
}

// -------------------------------------------------- gate/up GEMM + GLU ------
// hbuf layout: [e][mtile][16 x 2048] tiles, XOR-swizzled per swz_elem(), so a
// linear TDM copy of one 64 KB slab lands a conflict-free LDS image.
__global__ __launch_bounds__(256) void gateup_kernel(
    const float* __restrict__ x,
    const float* __restrict__ gblk, const float* __restrict__ gscl,
    const float* __restrict__ gbias,
    const float* __restrict__ ublk, const float* __restrict__ uscl,
    const float* __restrict__ ubias,
    const int* __restrict__ cnt, const int* __restrict__ tokidx,
    __bf16* __restrict__ hbuf) {
  int e = blockIdx.x >> 6;        // T_/16 = 64 m-tiles per expert
  int mtile = blockIdx.x & 63;
  int c = cnt[e];
  if (mtile * 16 >= c) return;

  __shared__ __align__(16) __bf16 Ax[16 * H_];  // 64 KB swizzled A tile
  {
    unsigned* axp = (unsigned*)Ax;
    for (int idx = threadIdx.x; idx < 16 * (H_ / 2); idx += 256) {
      int m = idx >> 10, hp = idx & (H_ / 2 - 1);
      int slot = mtile * 16 + m;
      int t = tokidx[e * T_ + (slot < c ? slot : c - 1)];
      float2 xv = *(const float2*)(x + (size_t)t * H_ + 2 * hp);
      int sw = (m << 10) | ((((hp >> 2) ^ m) << 2) | (hp & 3));
      axp[sw] = cvt_pk_bf16(xv.x, xv.y);
    }
  }
  __syncthreads();

  int wave = threadIdx.x >> 5, lane = threadIdx.x & 31;
  int kh = lane >> 4, ln = lane & 15;
  int i0 = blockIdx.y * 256 + wave * 32;

  // Hoisted per-lane row base pointers for the two N tiles.
  const float *grow[2], *urow[2], *gsp[2], *usp[2];
#pragma unroll
  for (int j = 0; j < 2; ++j) {
    size_t row = (size_t)(e * I_ + i0 + j * 16 + ln);
    grow[j] = gblk + row * H_;
    urow[j] = ublk + row * H_;
    gsp[j] = gscl + row * NBLK;
    usp[j] = uscl + row * NBLK;
  }
  const uint4* abase = (const uint4*)Ax;

  v8f ag[2], au[2];
#pragma unroll
  for (int j = 0; j < 2; ++j)
#pragma unroll
    for (int r = 0; r < 8; ++r) { ag[j][r] = 0.f; au[j][r] = 0.f; }

#pragma unroll 1
  for (int k0 = 0; k0 < H_; k0 += 32) {
    ABfrag a;
    load_a_frag_sw(abase, ln, k0, kh, a);
#pragma unroll
    for (int j = 0; j < 2; ++j) {
      ABfrag b;
      load_b_frag(grow[j] + k0, kh, gsp[j][k0 >> 5], b);
      ag[j] = __builtin_amdgcn_wmma_f32_16x16x32_bf16(
          false, a.bf, false, b.bf, (short)0, ag[j], false, false);
      ABfrag bu;
      load_b_frag(urow[j] + k0, kh, usp[j][k0 >> 5], bu);
      au[j] = __builtin_amdgcn_wmma_f32_16x16x32_bf16(
          false, a.bf, false, bu.bf, (short)0, au[j], false, false);
    }
  }

  __bf16* slab = hbuf + (size_t)(e * 64 + mtile) * (16 * 2048);
#pragma unroll
  for (int j = 0; j < 2; ++j) {
    int i = i0 + j * 16 + ln;                 // C/D: lane&15 is the N column
    float gb = gbias[e * I_ + i], ub = ubias[e * I_ + i];
#pragma unroll
    for (int r = 0; r < 8; ++r) {
      int m = r + 8 * kh;                     // C/D: VGPR r is row m (+8 hi half)
      float g = fminf(ag[j][r] + gb, LIMIT_);
      float u = fminf(fmaxf(au[j][r] + ub, -LIMIT_), LIMIT_);
      float glu = g * (1.f / (1.f + __expf(-ALPHA_ * g)));
      slab[swz_elem(m, i)] = (__bf16)((u + 1.f) * glu);
    }
  }
}

// ----------------------------------------- down GEMM (TDM-staged A tile) ----
__global__ __launch_bounds__(256) void down_kernel(
    const __bf16* __restrict__ hbuf,
    const float* __restrict__ dblk, const float* __restrict__ dscl,
    const float* __restrict__ dbias,
    const int* __restrict__ cnt, __bf16* __restrict__ obuf) {
  int e = blockIdx.x >> 6, mtile = blockIdx.x & 63;
  if (mtile * 16 >= cnt[e]) return;

  __shared__ __align__(16) __bf16 Ah[16 * I_];  // 64 KB pre-swizzled slab
  if (threadIdx.x < 32) {
    // Tensor Data Mover: 1D tile, 32768 x 2-byte elements, global -> LDS.
    uint64_t g =
        (uint64_t)(uintptr_t)(hbuf + (size_t)(e * 64 + mtile) * (16 * 2048));
    unsigned ldsoff = (unsigned)(uintptr_t)(void*)Ah;
    u32x4 g0;
    g0[0] = 1u;                                        // count=1, user desc
    g0[1] = ldsoff;                                    // lds_addr
    g0[2] = (unsigned)g;                               // global_addr[31:0]
    g0[3] = (unsigned)((g >> 32) & 0x1FFFFFFu) | (2u << 30);  // [56:32], type=2
    i32x8 g1;
#pragma unroll
    for (int q = 0; q < 8; ++q) g1[q] = 0;
    g1[0] = (1 << 16);                 // data_size = 2 bytes, no multicast
    g1[1] = (int)(32768u << 16);       // tensor_dim0 = 32768 (bits 79:48 lo)
    g1[3] = (int)(32768u << 16);       // tile_dim0   = 32768 (bits 127:112)
    g1[5] = 32768;                     // tensor_dim0_stride = 32768
    i32x4 g2 = {0, 0, 0, 0}, g3 = {0, 0, 0, 0};
    i32x8 g4;
#pragma unroll
    for (int q = 0; q < 8; ++q) g4[q] = 0;
    __builtin_amdgcn_tensor_load_to_lds(g0, g1, g2, g3, g4, 0);
    __builtin_amdgcn_s_wait_tensorcnt(0);
  }
  __syncthreads();

  int wave = threadIdx.x >> 5, lane = threadIdx.x & 31;
  int kh = lane >> 4, ln = lane & 15;
  int h0 = blockIdx.y * 256 + wave * 32;

  const float *drow[2], *dsp[2];
#pragma unroll
  for (int j = 0; j < 2; ++j) {
    size_t row = (size_t)(e * H_ + h0 + j * 16 + ln);
    drow[j] = dblk + row * I_;
    dsp[j] = dscl + row * NBLK;
  }
  const uint4* abase = (const uint4*)Ah;

  v8f ac[2];
#pragma unroll
  for (int j = 0; j < 2; ++j)
#pragma unroll
    for (int r = 0; r < 8; ++r) ac[j][r] = 0.f;

#pragma unroll 1
  for (int k0 = 0; k0 < I_; k0 += 32) {
    ABfrag a;
    load_a_frag_sw(abase, ln, k0, kh, a);
#pragma unroll
    for (int j = 0; j < 2; ++j) {
      ABfrag b;
      load_b_frag(drow[j] + k0, kh, dsp[j][k0 >> 5], b);
      ac[j] = __builtin_amdgcn_wmma_f32_16x16x32_bf16(
          false, a.bf, false, b.bf, (short)0, ac[j], false, false);
    }
  }

#pragma unroll
  for (int j = 0; j < 2; ++j) {
    int h = h0 + j * 16 + ln;
    float db = dbias[e * H_ + h];
#pragma unroll
    for (int r = 0; r < 8; ++r) {
      int slot = mtile * 16 + r + 8 * kh;
      obuf[(size_t)(e * T_ + slot) * H_ + h] = (__bf16)(ac[j][r] + db);
    }
  }
}

// --------------------------------------------------------------- combine ----
__global__ __launch_bounds__(256) void combine_kernel(
    const __bf16* __restrict__ obuf, const float* __restrict__ scores,
    const int* __restrict__ slotof, float* __restrict__ out) {
  int t = blockIdx.x;
  for (int h = threadIdx.x; h < H_; h += 256) {
    float acc = 0.f;
#pragma unroll
    for (int e = 0; e < E_; ++e) {       // fixed order -> deterministic sum
      float s = scores[(size_t)t * E_ + e];
      if (s > 0.f) {
        int slot = slotof[t * E_ + e];
        acc += s * (float)obuf[(size_t)(e * T_ + slot) * H_ + h];
      }
    }
    out[(size_t)t * H_ + h] = acc;
  }
}

// ---------------------------------------------------------------- launch ----
extern "C" void kernel_launch(void* const* d_in, const int* in_sizes, int n_in,
                              void* d_out, int out_size, void* d_ws, size_t ws_size,
                              hipStream_t stream) {
  (void)in_sizes; (void)n_in; (void)out_size; (void)ws_size;
  const float* x     = (const float*)d_in[0];
  const float* rw    = (const float*)d_in[1];
  const float* rb    = (const float*)d_in[2];
  const float* gblk  = (const float*)d_in[3];
  const float* gscl  = (const float*)d_in[4];
  const float* gbias = (const float*)d_in[5];
  const float* ublk  = (const float*)d_in[6];
  const float* uscl  = (const float*)d_in[7];
  const float* ubias = (const float*)d_in[8];
  const float* dblk  = (const float*)d_in[9];
  const float* dscl  = (const float*)d_in[10];
  const float* dbias = (const float*)d_in[11];
  float* out = (float*)d_out;

  char* ws = (char*)d_ws;
  float* scores = (float*)ws;            ws += (size_t)T_ * E_ * 4;
  int* cnt      = (int*)ws;              ws += 64 * 4;
  int* tokidx   = (int*)ws;              ws += (size_t)E_ * T_ * 4;
  int* slotof   = (int*)ws;              ws += (size_t)T_ * E_ * 4;
  uintptr_t p = ((uintptr_t)ws + 255) & ~(uintptr_t)255;
  __bf16* hbuf = (__bf16*)p;             p += (size_t)E_ * T_ * I_ * 2;
  __bf16* obuf = (__bf16*)p;             // + E_*T_*H_*2 bytes

  router_kernel<<<T_ / 8, 256, 0, stream>>>(x, rw, rb, scores);
  compact_kernel<<<1, 16, 0, stream>>>(scores, cnt, tokidx, slotof);
  gateup_kernel<<<dim3(E_ * (T_ / 16), I_ / 256), 256, 0, stream>>>(
      x, gblk, gscl, gbias, ublk, uscl, ubias, cnt, tokidx, hbuf);
  down_kernel<<<dim3(E_ * (T_ / 16), H_ / 256), 256, 0, stream>>>(
      hbuf, dblk, dscl, dbias, cnt, obuf);
  combine_kernel<<<T_, 256, 0, stream>>>(obuf, scores, slotof, out);
}
